// LaplacianModel_62079457296719
// MI455X (gfx1250) — matrix operand order repaired and checked by reading, not compiled, not run on hardware
//
#include <hip/hip_runtime.h>
#include <math.h>

typedef __attribute__((ext_vector_type(16))) _Float16 v16h;
typedef __attribute__((ext_vector_type(8)))  _Float16 v8h;
typedef __attribute__((ext_vector_type(8)))  float    v8f;

#define B_CENTERS   32768
#define N_STAG      2048383
#define N_FULL      2097152
#define ROWS_PER_BLOCK 64
#define MLP_THREADS 128
#define N_ROWS      (17 * B_CENTERS)          // 557056 = 8704 * 64
#define MLP_BLOCKS  (N_ROWS / ROWS_PER_BLOCK) // 8704
#define LOSS_BLOCKS 64

// ---------------------------------------------------------------------------
// Fused MLP kernel: gather -> L1 (VALU) -> L2,L3 (WMMA f16->f32) -> L4 (shfl)
// point p = t*B + b ; t=0: DATAX[idx], t=1..8: STAGX[stag], t=9..16: DATAX[neig]
// ---------------------------------------------------------------------------
__global__ void __launch_bounds__(MLP_THREADS)
mlp_kernel(const float* __restrict__ DATAX, const float* __restrict__ STAGX,
           const float* __restrict__ W1, const float* __restrict__ b1,
           const float* __restrict__ W2, const float* __restrict__ b2,
           const float* __restrict__ W3, const float* __restrict__ b3,
           const float* __restrict__ W4, const float* __restrict__ b4,
           const int* __restrict__ cidx, float* __restrict__ out)
{
    // B-matrix fragments, pre-swizzled to WMMA lane layout: 4 kc * 8 nt frags,
    // each frag = 32 lanes * 16 f16 (lane: n = lane&15, K = kc*32 + (i>>3)*16 + (lane>>4)*8 + (i&7))
    __shared__ __align__(32) _Float16 Wfrag[16384];                 // 32 KB
    __shared__ __align__(32) _Float16 hbuf[ROWS_PER_BLOCK * 128];   // 16 KB

    const int tid  = threadIdx.x;
    const int lane = tid & 31;
    const int wv   = tid >> 5;        // wave 0..3, owns rows [16*wv, 16*wv+16)
    const int half = lane >> 4;
    const int ml   = lane & 15;

    // ---- Phase 1: gather + layer 1 (two threads per row, 64 k each) ----
    {
        const int row_local = tid >> 1;
        const int kh = (tid & 1) * 64;
        const int p = blockIdx.x * ROWS_PER_BLOCK + row_local;
        const int b = p & (B_CENTERS - 1);
        const int t = p >> 15;
        const int idx = cidx[b];
        const float* src;
        if (t == 0) {
            src = DATAX + 3 * idx;
        } else {
            const int ix = idx >> 14, iy = (idx >> 7) & 127, iz = idx & 127;
            if (t <= 8) {
                const int s = t - 1;
                const int dx = (s & 4) ? 0 : -1;
                const int dy = (s & 2) ? 0 : -1;
                const int dz = (s & 1) ? 0 : -1;
                int f = (ix + dx) * 16129 + (iy + dy) * 127 + (iz + dz);
                f = min(max(f, 0), N_STAG - 1);
                src = STAGX + 3 * f;
            } else {
                const int s = t - 9;
                const int dx = (s & 4) ? 1 : -1;
                const int dy = (s & 2) ? 1 : -1;
                const int dz = (s & 1) ? 1 : -1;
                int f = (ix + dx) * 16129 + (iy + dy) * 127 + (iz + dz);
                f = min(max(f, 0), N_FULL - 1);
                src = DATAX + 3 * f;
            }
        }
        const float x0 = src[0], x1 = src[1], x2 = src[2];
        for (int k = kh; k < kh + 64; ++k) {
            float a = b1[k] + x0 * W1[k] + x1 * W1[128 + k] + x2 * W1[256 + k];
            hbuf[row_local * 128 + k] = (_Float16)tanhf(a);
        }
    }

    // ---- Stage W2 fragments into LDS (swizzled) ----
    for (int e = tid; e < 16384; e += MLP_THREADS) {
        const int fragIdx = e >> 9, inFrag = e & 511;
        const int fl = inFrag >> 4, ii = inFrag & 15;
        const int kcc = fragIdx >> 3, nt = fragIdx & 7;
        const int k   = kcc * 32 + (ii >> 3) * 16 + (fl >> 4) * 8 + (ii & 7);
        const int col = nt * 16 + (fl & 15);
        Wfrag[e] = (_Float16)W2[k * 128 + col];
    }
    __syncthreads();

    const v8f vzero = {0.f, 0.f, 0.f, 0.f, 0.f, 0.f, 0.f, 0.f};
    const _Float16* hrow = &hbuf[(wv * 16 + ml) * 128];

    // ---- Layer 2: 16x128 strip per wave, 8 n-tiles x 4 k-chunks ----
    {
        v8f acc[8];
        #pragma unroll
        for (int nt = 0; nt < 8; ++nt) acc[nt] = vzero;
        #pragma unroll
        for (int kc = 0; kc < 4; ++kc) {
            union { v16h v; v8h h[2]; } ua;
            ua.h[0] = *(const v8h*)(hrow + kc * 32 + half * 8);       // K = base..base+7
            ua.h[1] = *(const v8h*)(hrow + kc * 32 + 16 + half * 8);  // K = base+16..+23
            #pragma unroll
            for (int nt = 0; nt < 8; ++nt) {
                v16h bf = *(const v16h*)&Wfrag[((kc * 8 + nt) * 32 + lane) * 16];
                acc[nt] = __builtin_amdgcn_wmma_f32_16x16x32_f16(
                    false, ua.v, false, bf, (short)0, acc[nt], false, false);
            }
        }
        #pragma unroll
        for (int nt = 0; nt < 8; ++nt) {
            const int col = nt * 16 + ml;
            const float bb = b2[col];
            #pragma unroll
            for (int r = 0; r < 8; ++r) {
                // D layout: lane(n=col), VGPR r -> row m = r + 8*half
                hbuf[(wv * 16 + r + 8 * half) * 128 + col] =
                    (_Float16)tanhf(acc[nt][r] + bb);
            }
        }
    }
    __syncthreads();

    // ---- Stage W3 fragments (reuse Wfrag) ----
    for (int e = tid; e < 16384; e += MLP_THREADS) {
        const int fragIdx = e >> 9, inFrag = e & 511;
        const int fl = inFrag >> 4, ii = inFrag & 15;
        const int kcc = fragIdx >> 3, nt = fragIdx & 7;
        const int k   = kcc * 32 + (ii >> 3) * 16 + (fl >> 4) * 8 + (ii & 7);
        const int col = nt * 16 + (fl & 15);
        Wfrag[e] = (_Float16)W3[k * 128 + col];
    }
    __syncthreads();

    // ---- Layer 3 (WMMA) fused with layer 4 dot product ----
    {
        v8f acc[8];
        #pragma unroll
        for (int nt = 0; nt < 8; ++nt) acc[nt] = vzero;
        #pragma unroll
        for (int kc = 0; kc < 4; ++kc) {
            union { v16h v; v8h h[2]; } ua;
            ua.h[0] = *(const v8h*)(hrow + kc * 32 + half * 8);
            ua.h[1] = *(const v8h*)(hrow + kc * 32 + 16 + half * 8);
            #pragma unroll
            for (int nt = 0; nt < 8; ++nt) {
                v16h bf = *(const v16h*)&Wfrag[((kc * 8 + nt) * 32 + lane) * 16];
                acc[nt] = __builtin_amdgcn_wmma_f32_16x16x32_f16(
                    false, ua.v, false, bf, (short)0, acc[nt], false, false);
            }
        }
        float psum[8];
        #pragma unroll
        for (int r = 0; r < 8; ++r) psum[r] = 0.f;
        #pragma unroll
        for (int nt = 0; nt < 8; ++nt) {
            const int col = nt * 16 + ml;
            const float bb = b3[col];
            const float w4 = W4[col];
            #pragma unroll
            for (int r = 0; r < 8; ++r)
                psum[r] += tanhf(acc[nt][r] + bb) * w4;
        }
        const float b4v = b4[0];
        #pragma unroll
        for (int r = 0; r < 8; ++r) {
            float s = psum[r];
            s += __shfl_xor(s, 1);
            s += __shfl_xor(s, 2);
            s += __shfl_xor(s, 4);
            s += __shfl_xor(s, 8);   // sum over the 16-lane half-group
            if (ml == r) {
                const int p = blockIdx.x * ROWS_PER_BLOCK + wv * 16 + r + 8 * half;
                out[p] = s + b4v;
            }
        }
    }
}

// ---------------------------------------------------------------------------
// Loss reduction: deterministic per-block partials (no float atomics)
// ---------------------------------------------------------------------------
__global__ void __launch_bounds__(256)
loss_kernel(const float* __restrict__ mlp_out, const float* __restrict__ DATAF,
            const float* __restrict__ LAPLF, const int* __restrict__ cidx,
            float* __restrict__ partials)
{
    float se = 0.f, sm = 0.f, sc = 0.f;
    for (int b = blockIdx.x * blockDim.x + threadIdx.x; b < B_CENTERS;
         b += gridDim.x * blockDim.x) {
        const int idx = cidx[b];
        const int ix = idx >> 14, iy = (idx >> 7) & 127, iz = idx & 127;
        se += fabsf(mlp_out[b] - DATAF[idx]);
        const bool interior = (ix >= 1) && (ix < 127) && (iy >= 1) && (iy < 127)
                              && (iz >= 1) && (iz < 127);
        if (interior) {
            float ps[8], pn[8];
            #pragma unroll
            for (int k = 0; k < 8; ++k) {
                ps[k] = mlp_out[(1 + k) * B_CENTERS + b];
                pn[k] = mlp_out[(9 + k) * B_CENTERS + b];
            }
            const float td = 1.5f; // 3 * 2 * 0.5^2
            float lap =
                fabsf(pn[7] + pn[0] - ps[7] - ps[0]) / ((pn[7] + pn[0] + ps[7] + ps[0]) * td)
              + fabsf(pn[3] + pn[4] - ps[3] - ps[4]) / ((pn[3] + pn[4] + ps[3] + ps[4]) * td)
              + fabsf(pn[5] + pn[2] - ps[5] - ps[2]) / ((pn[5] + pn[2] + ps[5] + ps[2]) * td);
            const float d = LAPLF[idx] - lap;
            sm += d * d;
            sc += 1.f;
        }
    }
    __shared__ float red0[256], red1[256], red2[256];
    const int t = threadIdx.x;
    red0[t] = se; red1[t] = sm; red2[t] = sc;
    __syncthreads();
    for (int s = 128; s > 0; s >>= 1) {
        if (t < s) { red0[t] += red0[t + s]; red1[t] += red1[t + s]; red2[t] += red2[t + s]; }
        __syncthreads();
    }
    if (t == 0) {
        partials[blockIdx.x]                   = red0[0];
        partials[LOSS_BLOCKS + blockIdx.x]     = red1[0];
        partials[2 * LOSS_BLOCKS + blockIdx.x] = red2[0];
    }
}

__global__ void __launch_bounds__(64)
finalize_kernel(const float* __restrict__ partials, float* __restrict__ out)
{
    __shared__ float r0[64], r1[64], r2[64];
    const int t = threadIdx.x;
    r0[t] = partials[t];
    r1[t] = partials[LOSS_BLOCKS + t];
    r2[t] = partials[2 * LOSS_BLOCKS + t];
    __syncthreads();
    for (int s = 32; s > 0; s >>= 1) {
        if (t < s) { r0[t] += r0[t + s]; r1[t] += r1[t + s]; r2[t] += r2[t + s]; }
        __syncthreads();
    }
    if (t == 0)
        out[0] = r0[0] / (float)B_CENTERS + r1[0] / fmaxf(r2[0], 1.0f);
}

// ---------------------------------------------------------------------------
extern "C" void kernel_launch(void* const* d_in, const int* in_sizes, int n_in,
                              void* d_out, int out_size, void* d_ws, size_t ws_size,
                              hipStream_t stream) {
    const float* DATAX = (const float*)d_in[0];
    const float* DATAF = (const float*)d_in[1];
    const float* STAGX = (const float*)d_in[2];
    const float* LAPLF = (const float*)d_in[3];
    const float* W1    = (const float*)d_in[4];
    const float* b1    = (const float*)d_in[5];
    const float* W2    = (const float*)d_in[6];
    const float* b2    = (const float*)d_in[7];
    const float* W3    = (const float*)d_in[8];
    const float* b3    = (const float*)d_in[9];
    const float* W4    = (const float*)d_in[10];
    const float* b4    = (const float*)d_in[11];
    const int*   cidx  = (const int*)d_in[12];

    float* ws       = (float*)d_ws;
    float* partials = ws;          // 3 * LOSS_BLOCKS floats
    float* mlp_out  = ws + 256;    // 17*B floats

    mlp_kernel<<<MLP_BLOCKS, MLP_THREADS, 0, stream>>>(
        DATAX, STAGX, W1, b1, W2, b2, W3, b3, W4, b4, cidx, mlp_out);
    loss_kernel<<<LOSS_BLOCKS, 256, 0, stream>>>(
        mlp_out, DATAF, LAPLF, cidx, partials);
    finalize_kernel<<<1, 64, 0, stream>>>(partials, (float*)d_out);
}
